// WindowAttention_41738492182688
// MI455X (gfx1250) — compile-verified, matrix operand order checked by reading, not run on hardware
//
#include <hip/hip_runtime.h>
#include <hip/hip_bf16.h>
#include <math.h>

// ---------------------------------------------------------------------------
// WindowAttention + progressive focusing for gfx1250 (MI455X).
//   K0: proj_w f32 -> f16 (one shot)
//   K1: per-query wave32 sparse attention, writes f16 activations
//   K2: dense projection; TDM tensor_load_to_lds stages weight slab in LDS,
//       v_wmma_f32_16x16x32_f16; B fragments via ds_load_b128 (addrspace 3).
// ---------------------------------------------------------------------------

typedef __attribute__((ext_vector_type(16))) _Float16     v16h;
typedef __attribute__((ext_vector_type(8)))  _Float16     h8;
typedef __attribute__((ext_vector_type(8)))  float        v8f;
typedef __attribute__((ext_vector_type(4)))  unsigned int u32x4;
typedef __attribute__((ext_vector_type(8)))  int          i32x8;
typedef __attribute__((ext_vector_type(4)))  int          i32x4;

// explicit LDS (addrspace 3) pointer to 8 x f16 — keeps B-fragment reads on
// the DS pipe (ds_load_b128) instead of FLAT.
typedef const h8 __attribute__((address_space(3)))* lds_h8p;

#define B_     32
#define N_     256
#define H_     6
#define HD_    32
#define C_     192           // H_*HD_
#define TKIN   64
#define TKOUT  32
#define SCALE_ 0.1767766952966369f   // 32^-0.5
#define EPS_   1e-20f

#define MTILES ((B_ * N_) / 16)   // 512
#define NTILES (C_ / 16)          // 12

// ---------------------------------------------------------------------------
// Kernel 0: proj_w [C,C] f32 -> f16
// ---------------------------------------------------------------------------
__global__ __launch_bounds__(256)
void w_to_f16_kernel(const float* __restrict__ w, _Float16* __restrict__ wh)
{
    const int i = blockIdx.x * 256 + threadIdx.x;
    if (i < C_ * C_) wh[i] = (_Float16)w[i];
}

// ---------------------------------------------------------------------------
// Kernel 1: one wave (32 threads) per (b, h, query i).
// ---------------------------------------------------------------------------
__global__ __launch_bounds__(32)
void pfa_attn_kernel(const float* __restrict__ qkvp,
                     const float* __restrict__ pfa_values,
                     const int*   __restrict__ pfa_index,
                     const int*   __restrict__ rpi,
                     const float* __restrict__ rpb_table,
                     _Float16* __restrict__ xh,       // [B,N,C] f16 activations
                     float* __restrict__ out_topv,    // [B,H,N,TKOUT]
                     float* __restrict__ out_idx)     // [B,H,N,TKOUT]
{
    __shared__ float s_q[HD_];
    __shared__ float s_att[TKIN];
    __shared__ int   s_idx[TKIN];
    __shared__ float s_tv[TKOUT];
    __shared__ int   s_ti[TKOUT];

    const int lane = threadIdx.x;            // 0..31
    const int bid  = blockIdx.x;             // b*H*N + h*N + i
    const int i    = bid % N_;
    const int h    = (bid / N_) % H_;
    const int b    = bid / (N_ * H_);

    const size_t rowstride = 4 * C_;         // qkvp row: [4][H][HD]

    const float* qrow = qkvp + (size_t)(b * N_ + i) * rowstride + h * HD_;
    s_q[lane] = qrow[lane] * SCALE_;
    __syncthreads();

    const size_t pbase = ((size_t)(b * H_ + h) * N_ + i) * TKIN;

    float av[2];
    int   ki[2];
#pragma unroll
    for (int s = 0; s < 2; ++s) {
        const int t    = lane + s * 32;
        const int kidx = pfa_index[pbase + t];
        ki[s] = kidx;
        const float4* k4 = (const float4*)(qkvp + (size_t)(b * N_ + kidx) * rowstride
                                           + C_ + h * HD_);
        float dot = 0.f;
#pragma unroll
        for (int d4 = 0; d4 < HD_ / 4; ++d4) {
            const float4 kv = k4[d4];
            dot += s_q[4 * d4 + 0] * kv.x + s_q[4 * d4 + 1] * kv.y +
                   s_q[4 * d4 + 2] * kv.z + s_q[4 * d4 + 3] * kv.w;
        }
        const int bi = rpi[(size_t)i * N_ + kidx];
        av[s] = dot + rpb_table[(size_t)bi * H_ + h];
    }

    // softmax over 64 candidates (2 per lane), wave32 shuffle tree
    float m = fmaxf(av[0], av[1]);
#pragma unroll
    for (int off = 16; off > 0; off >>= 1) m = fmaxf(m, __shfl_xor(m, off, 32));
    const float e0 = __expf(av[0] - m);
    const float e1 = __expf(av[1] - m);
    float ssum = e0 + e1;
#pragma unroll
    for (int off = 16; off > 0; off >>= 1) ssum += __shfl_xor(ssum, off, 32);
    const float sinv = 1.f / ssum;

    // progressive focusing renorm
    const float w0 = e0 * sinv * pfa_values[pbase + lane];
    const float w1 = e1 * sinv * pfa_values[pbase + lane + 32];
    float dsum = w0 + w1;
#pragma unroll
    for (int off = 16; off > 0; off >>= 1) dsum += __shfl_xor(dsum, off, 32);
    const float dinv = 1.f / (dsum + EPS_);
    const float f0 = (w0 + EPS_) * dinv;
    const float f1 = (w1 + EPS_) * dinv;

    s_att[lane]      = f0;  s_idx[lane]      = ki[0];
    s_att[lane + 32] = f1;  s_idx[lane + 32] = ki[1];
    __syncthreads();

    // exact top-32 with top_k tie-break (earlier position wins)
#pragma unroll
    for (int s = 0; s < 2; ++s) {
        const int   t  = lane + s * 32;
        const float mv = s_att[t];
        int rank = 0;
        for (int u = 0; u < TKIN; ++u) {
            const float vu = s_att[u];
            rank += (vu > mv) || (vu == mv && u < t);
        }
        if (rank < TKOUT) { s_tv[rank] = mv; s_ti[rank] = s_idx[t]; }
    }
    __syncthreads();

    const size_t obase = ((size_t)(b * H_ + h) * N_ + i) * TKOUT;
    out_topv[obase + lane] = s_tv[lane];
    out_idx [obase + lane] = (float)s_ti[lane];

    // sparse AV over kept keys + LePE; lane = head channel
    float acc = qkvp[(size_t)(b * N_ + i) * rowstride + 3 * C_ + h * HD_ + lane];
    for (int r = 0; r < TKOUT; ++r) {
        acc += s_tv[r] *
               qkvp[(size_t)(b * N_ + s_ti[r]) * rowstride + 2 * C_ + h * HD_ + lane];
    }
    xh[(size_t)(b * N_ + i) * C_ + h * HD_ + lane] = (_Float16)acc;
}

// ---------------------------------------------------------------------------
// TDM: stage a [32 rows x 192 cols] f16 weight slab (12 KB) into LDS offset 0.
// D# per CDNA5 ISA 8.3/8.4: group0 {count=1, lds_addr, global_addr(57b),
// type=2}; group1 {data_size=2B, tensor_dim0=192, tensor_dim1=32,
// tile_dim0=192, tile_dim1=32, tensor_dim0_stride=192}. Groups 2/3 zero (2D).
// This toolchain uses the 6-arg builtin form (extra int32x8 before cpol).
// ---------------------------------------------------------------------------
__device__ __forceinline__ void tdm_stage_weights(const _Float16* gsrc,
                                                  unsigned lds_byte_off)
{
    const unsigned long long ga = (unsigned long long)(size_t)gsrc;

    u32x4 g0;
    g0[0] = 1u;                                            // count=1, user D#
    g0[1] = lds_byte_off;                                  // lds_addr (bytes)
    g0[2] = (unsigned)(ga & 0xFFFFFFFFull);                // global_addr[31:0]
    g0[3] = (unsigned)((ga >> 32) & 0x01FFFFFFull)         // global_addr[56:32]
          | (2u << 30);                                    // type=2 ("image")

    i32x8 g1;
    g1[0] = (int)(1u << 16);        // workgroup_mask=0, data_size=1 (2 bytes)
    g1[1] = (int)(192u << 16);      // tensor_dim0[15:0]=192 @bits[63:48]
    g1[2] = (int)(32u  << 16);      // tensor_dim1[15:0]=32  @bits[95:80]
    g1[3] = (int)(192u << 16);      // tile_dim0=192         @bits[127:112]
    g1[4] = (int)(32u);             // tile_dim1=32, tile_dim2=0
    g1[5] = (int)(192u);            // tensor_dim0_stride[31:0]=192
    g1[6] = 0;                      // stride hi / dim1_stride lo
    g1[7] = 0;

    i32x4 gz4  = {0, 0, 0, 0};
    i32x8 gz8  = {0, 0, 0, 0, 0, 0, 0, 0};

    __builtin_amdgcn_tensor_load_to_lds(g0, g1, gz4, gz4, gz8, 0);
}

// ---------------------------------------------------------------------------
// Kernel 2: out[r, j] = sum_c x[r,c] * w[j,c] + b[j], f16 WMMA.
// Block = 2 N-tiles (32 cols, weights in LDS via TDM) x 8 M-tiles (1/wave).
// 16-bit fragment layout (ISA 7.12.2): lanes 0-15 hold K {0..7,16..23},
// lanes 16-31 hold K {8..15,24..31}; f32 D: m = r + (lane>=16 ? 8:0), n=lane&15.
// ---------------------------------------------------------------------------
__global__ __launch_bounds__(256)
void pfa_proj_wmma_kernel(const _Float16* __restrict__ xh,
                          const _Float16* __restrict__ wh,
                          const float*    __restrict__ proj_b,
                          float* __restrict__ out)
{
    // Sole shared object in this kernel -> LDS offset 0, matching the D#'s
    // lds_addr. Written only by the TDM engine; read below via explicit
    // addrspace(3) offsets so the loads stay on the DS pipe.
    __shared__ _Float16 s_w[32 * C_];   // 12 KB

    const int wv   = threadIdx.x >> 5;          // wave 0..7 -> M-tile
    const int lane = threadIdx.x & 31;
    const int cb   = blockIdx.x % (NTILES / 2); // column slab: 32 cols
    const int mbk  = blockIdx.x / (NTILES / 2); // 8 M-tiles per block

    // one TDM DMA per workgroup stages the 32 weight rows this slab needs
    if (wv == 0) {
        tdm_stage_weights(wh + (size_t)cb * 32 * C_, 0u);
        __builtin_amdgcn_s_wait_tensorcnt(0);
    }
    __syncthreads();

    // Anchor the LDS allocation (compiler would otherwise drop the unwritten
    // array) and forbid motion of the DS reads across the DMA/barrier.
    asm volatile("" : : "v"((void*)s_w) : "memory");

    const int row0 = (mbk * 8 + wv) * 16;
    const int l15  = lane & 15;
    const bool hi  = lane >= 16;

    const _Float16* xr = xh + (size_t)(row0 + l15) * C_;

    // byte offsets of the two weight rows this lane reads (slab base = 0)
    const unsigned wrow0 = (unsigned)l15 * (C_ * 2);
    const unsigned wrow1 = (unsigned)(l15 + 16) * (C_ * 2);

    v8f acc0 = {};
    v8f acc1 = {};
#pragma unroll
    for (int kk = 0; kk < C_; kk += 32) {
        const int b0 = kk + (hi ? 8 : 0);
        const int b1 = kk + (hi ? 24 : 16);

        const h8 alo = *(const h8*)(xr + b0);
        const h8 ahi = *(const h8*)(xr + b1);
        const v16h a = __builtin_shufflevector(alo, ahi,
                           0,1,2,3,4,5,6,7,8,9,10,11,12,13,14,15);

        const h8 w0lo = *(lds_h8p)(wrow0 + (unsigned)(b0 * 2));
        const h8 w0hi = *(lds_h8p)(wrow0 + (unsigned)(b1 * 2));
        const v16h bf0 = __builtin_shufflevector(w0lo, w0hi,
                           0,1,2,3,4,5,6,7,8,9,10,11,12,13,14,15);

        const h8 w1lo = *(lds_h8p)(wrow1 + (unsigned)(b0 * 2));
        const h8 w1hi = *(lds_h8p)(wrow1 + (unsigned)(b1 * 2));
        const v16h bf1 = __builtin_shufflevector(w1lo, w1hi,
                           0,1,2,3,4,5,6,7,8,9,10,11,12,13,14,15);

        acc0 = __builtin_amdgcn_wmma_f32_16x16x32_f16(
                   false, a, false, bf0, (short)0, acc0, false, false);
        acc1 = __builtin_amdgcn_wmma_f32_16x16x32_f16(
                   false, a, false, bf1, (short)0, acc1, false, false);
    }

    const int   mb    = hi ? 8 : 0;
    const int   col0  = cb * 32 + l15;
    const float bias0 = proj_b[col0];
    const float bias1 = proj_b[col0 + 16];
#pragma unroll
    for (int r = 0; r < 8; ++r) {
        out[(size_t)(row0 + mb + r) * C_ + col0]      = acc0[r] + bias0;
        out[(size_t)(row0 + mb + r) * C_ + col0 + 16] = acc1[r] + bias1;
    }
}

// ---------------------------------------------------------------------------
// Launch. d_out = [ x (B*N*C) | topv (B*H*N*32) | new_index (B*H*N*32) ] f32.
// d_ws = [ xh f16 (B*N*C) | wh f16 (C*C) ].
// ---------------------------------------------------------------------------
extern "C" void kernel_launch(void* const* d_in, const int* in_sizes, int n_in,
                              void* d_out, int out_size, void* d_ws, size_t ws_size,
                              hipStream_t stream)
{
    const float* qkvp       = (const float*)d_in[0];
    const float* pfa_values = (const float*)d_in[1];
    const int*   pfa_index  = (const int*)  d_in[2];
    const int*   rpi        = (const int*)  d_in[3];
    const float* rpb_table  = (const float*)d_in[4];
    const float* proj_w     = (const float*)d_in[5];
    const float* proj_b     = (const float*)d_in[6];

    float* out_x    = (float*)d_out;
    float* out_topv = out_x    + (size_t)B_ * N_ * C_;
    float* out_idx  = out_topv + (size_t)B_ * H_ * N_ * TKOUT;

    _Float16* xh = (_Float16*)d_ws;                      // 3 MB
    _Float16* wh = xh + (size_t)B_ * N_ * C_;            // 72 KB

    // Stage 0: weights to f16
    w_to_f16_kernel<<<(C_ * C_ + 255) / 256, 256, 0, stream>>>(proj_w, wh);

    // Stage 1: sparse attention, one wave32 per (b, h, query)
    pfa_attn_kernel<<<B_ * H_ * N_, 32, 0, stream>>>(
        qkvp, pfa_values, pfa_index, rpi, rpb_table, xh, out_topv, out_idx);

    // Stage 2: WMMA projection; 6 column slabs x 64 M-blocks = 384 blocks
    pfa_proj_wmma_kernel<<<(NTILES / 2) * (MTILES / 8), 256, 0, stream>>>(
        xh, wh, proj_b, out_x);
}
